// CNNMode_Kernal_2_64776696758914
// MI455X (gfx1250) — matrix operand order repaired and checked by reading, not compiled, not run on hardware
//
#include <hip/hip_runtime.h>

// ---------------------------------------------------------------------------
// Problem constants
// ---------------------------------------------------------------------------
#define IN_CH   12
#define STEP    4
#define UPLAG   2048
#define FEAT    (UPLAG / STEP)        // 512
#define DIN     (IN_CH * FEAT)        // 6144
#define HIDDEN  1024
#define OUTPUT  256
#define BATCH   2048

#define KT1     (DIN / 32)            // 192 k-tiles for GEMM1
#define KT2     (HIDDEN / 32)         // 32  k-tiles for GEMM2

#define ABLK    (KT1 * 1024)          // byte stride between adjacent 16-row A subtiles
#define BBLK    (KT1 * 1024)          // byte stride between adjacent 16-col B subtiles

typedef __attribute__((ext_vector_type(16))) __bf16 v16bf;
typedef __attribute__((ext_vector_type(8)))  float  v8f;

union AFrag { v16bf v; unsigned int u[8]; };

__device__ __forceinline__ unsigned short f2bf(float f) {
  unsigned int u = __float_as_uint(f);
  unsigned int r = u + 0x7FFFu + ((u >> 16) & 1u);   // round-to-nearest-even
  return (unsigned short)(r >> 16);
}

// ---------------------------------------------------------------------------
// Pack a row-major f32 weight matrix W[K,N] into bf16 B-fragment blocks.
// Block (nt,kt) = 512 bf16 at ((nt*Kt + kt)*512); within a block, lane L owns
// 16 bf16 (32 B). ISA layout: lane = (n&15) + 16*hi, hi = (k>>3)&1 within the
// 32-deep tile; slot g = ((k&16)>>2) + ((k&7)>>1), low/high half = k even/odd.
// ---------------------------------------------------------------------------
__global__ __launch_bounds__(256)
void pack_B_kernel(const float* __restrict__ W, unsigned int* __restrict__ Bpk,
                   int K, int N) {
  int t = blockIdx.x * 256 + threadIdx.x;
  int total = (K >> 1) * N;
  if (t >= total) return;
  int n  = t % N;
  int k  = (t / N) * 2;
  int Kt = K >> 5;
  int kt = k >> 5;
  int hi = (k >> 3) & 1;
  int g  = ((k & 16) >> 2) + ((k & 7) >> 1);
  int nt = n >> 4;
  int L  = (n & 15) + 16 * hi;
  unsigned int dw = (unsigned int)f2bf(W[k * N + n])
                  | ((unsigned int)f2bf(W[(k + 1) * N + n]) << 16);
  Bpk[(nt * Kt + kt) * 256 + L * 8 + g] = dw;
}

// ---------------------------------------------------------------------------
// Fused conv (per-channel non-overlapping K=4 dot + bias) -> bf16 A-fragment
// blocks. Each thread computes an even (kcol,kcol+1) pair for one batch row:
// 32 contiguous bytes of x in (coalesced), one packed dword out.
// ISA 16-bit A 16x32 layout: lane = m + 16*hi, hi = (kin>>3)&1,
// slot g = ((kin&16)>>2) + ((kin&7)>>1), low/high = kin even/odd.
// ---------------------------------------------------------------------------
__global__ __launch_bounds__(256)
void conv_pack_kernel(const float* __restrict__ x, const float* __restrict__ W1,
                      const float* __restrict__ b1, unsigned int* __restrict__ Apk) {
  int t = blockIdx.x * 256 + threadIdx.x;
  if (t >= BATCH * (DIN >> 1)) return;
  int kp = t % (DIN >> 1);
  int b  = t / (DIN >> 1);
  int k  = kp * 2;
  int c  = k >> 9;                 // / FEAT
  int f  = k & (FEAT - 1);
  const float* xp = x + ((b * IN_CH + c) * UPLAG + f * STEP);
  float w0 = W1[c * 4 + 0], w1 = W1[c * 4 + 1];
  float w2 = W1[c * 4 + 2], w3 = W1[c * 4 + 3];
  float bb = b1[c];
  float v0 = xp[0]*w0 + xp[1]*w1 + xp[2]*w2 + xp[3]*w3 + bb;
  float v1 = xp[4]*w0 + xp[5]*w1 + xp[6]*w2 + xp[7]*w3 + bb;
  int mt  = b >> 4, m = b & 15;
  int kt  = k >> 5, kin = k & 31;
  int hi  = (kin >> 3) & 1;
  int g   = ((kin & 16) >> 2) + ((kin & 7) >> 1);
  int L   = m + 16 * hi;
  unsigned int dw = (unsigned int)f2bf(v0) | ((unsigned int)f2bf(v1) << 16);
  Apk[(mt * KT1 + kt) * 256 + L * 8 + g] = dw;
}

// ---------------------------------------------------------------------------
// GEMM1: h = relu(A[2048,6144] @ W2[6144,1024] + b2), bf16 in / f32 acc.
// One wave per 64x32 output tile (4M x 2N WMMA tiles): 8 in-place accumulators.
// Manual 2-tile ping-pong software pipeline: while tile 2t's 8 WMMAs execute,
// tile 2t+1's 6 fragments are already in flight (and vice versa) — no
// cur/next register copies, accumulators never move.
// ---------------------------------------------------------------------------
__global__ __launch_bounds__(32)
void gemm1_kernel(const unsigned short* __restrict__ Apk,
                  const unsigned short* __restrict__ Bpk,
                  const float* __restrict__ b2,
                  unsigned short* __restrict__ h) {
  const int lane = threadIdx.x;
  const int ntp  = blockIdx.x;      // pair of N tiles   (0..31)
  const int mtq  = blockIdx.y;      // quad of M tiles   (0..31)

  const char* pA = (const char*)Apk + (size_t)(4 * mtq) * ABLK + lane * 32;
  const char* pB = (const char*)Bpk + (size_t)(2 * ntp) * BBLK + lane * 32;

  v8f acc[4][2];
#pragma unroll
  for (int j = 0; j < 4; ++j)
#pragma unroll
    for (int i = 0; i < 2; ++i) acc[j][i] = {};

  v16bf a0[4], b0[2], a1[4], b1[2];

#define LOAD_FRAGS(av, bv, off)                                           \
  do {                                                                    \
    _Pragma("unroll")                                                     \
    for (int j = 0; j < 4; ++j) (av)[j] = *(const v16bf*)(pA + j * ABLK + (off)); \
    _Pragma("unroll")                                                     \
    for (int i = 0; i < 2; ++i) (bv)[i] = *(const v16bf*)(pB + i * BBLK + (off)); \
  } while (0)

#define DO_WMMA(av, bv)                                                   \
  do {                                                                    \
    _Pragma("unroll")                                                     \
    for (int j = 0; j < 4; ++j)                                           \
      _Pragma("unroll")                                                   \
      for (int i = 0; i < 2; ++i)                                         \
        acc[j][i] = __builtin_amdgcn_wmma_f32_16x16x32_bf16(              \
            false, (av)[j], false, (bv)[i], (short)0, acc[j][i], false, false); \
  } while (0)

  // Preload k-tile 0 into buffer 0.
  LOAD_FRAGS(a0, b0, 0);

  // Steady state: 95 iterations x 2 k-tiles; tiles 0..189 computed,
  // tile 190 left in buffer 0.
  for (int t = 0; t < (KT1 - 2) / 2; ++t) {
    LOAD_FRAGS(a1, b1, 1024);       // tile 2t+1
    DO_WMMA(a0, b0);                // tile 2t
    LOAD_FRAGS(a0, b0, 2048);       // tile 2t+2
    pA += 2048;
    pB += 2048;
    DO_WMMA(a1, b1);                // tile 2t+1
  }
  // Tail: tiles 190 and 191.
  LOAD_FRAGS(a1, b1, 1024);         // tile 191
  DO_WMMA(a0, b0);                  // tile 190
  DO_WMMA(a1, b1);                  // tile 191

#undef LOAD_FRAGS
#undef DO_WMMA

  // Epilogue: bias + ReLU + bf16 store. C layout: m = v + 8*hi, n = lane&15.
  int hi = lane >> 4, ln = lane & 15;
  float bias[2];
  int ncol[2];
#pragma unroll
  for (int i = 0; i < 2; ++i) {
    ncol[i] = (2 * ntp + i) * 16 + ln;
    bias[i] = b2[ncol[i]];
  }
#pragma unroll
  for (int j = 0; j < 4; ++j) {
#pragma unroll
    for (int v = 0; v < 8; ++v) {
      int m = (4 * mtq + j) * 16 + v + 8 * hi;
#pragma unroll
      for (int i = 0; i < 2; ++i)
        h[m * HIDDEN + ncol[i]] = f2bf(fmaxf(acc[j][i][v] + bias[i], 0.0f));
    }
  }
}

// ---------------------------------------------------------------------------
// GEMM2: out = h[2048,1024] @ W3[1024,256] + b3 (f32 out).
// A fragments gathered from row-major bf16 h via dword loads (fragment k-pairs
// are contiguous in memory); B from packed W3. One wave per 16x16 tile.
// ---------------------------------------------------------------------------
__global__ __launch_bounds__(32)
void gemm2_kernel(const unsigned short* __restrict__ h,
                  const unsigned short* __restrict__ Bpk,
                  const float* __restrict__ b3,
                  float* __restrict__ out) {
  const int lane = threadIdx.x;
  const int nt = blockIdx.x;        // 0..15
  const int mt = blockIdx.y;        // 0..127
  int hi = lane >> 4, ln = lane & 15;
  int m = mt * 16 + ln;             // A-fragment row for this lane

  const v16bf* Bb = (const v16bf*)Bpk;
  int ib = nt * KT2 * 32 + lane;

  v8f c = {};
  for (int kt = 0; kt < KT2; ++kt) {
    AFrag a;
#pragma unroll
    for (int g = 0; g < 8; ++g) {
      int kin = ((g < 4) ? (2 * g) : (16 + 2 * (g - 4))) + 8 * hi;   // even
      a.u[g] = *(const unsigned int*)(h + m * HIDDEN + kt * 32 + kin);
    }
    v16bf bv = Bb[ib + kt * 32];
    c = __builtin_amdgcn_wmma_f32_16x16x32_bf16(false, a.v, false, bv, (short)0, c, false, false);
  }

  float bias = b3[nt * 16 + ln];
#pragma unroll
  for (int v = 0; v < 8; ++v) {
    int mo = mt * 16 + v + 8 * hi;
    out[mo * OUTPUT + nt * 16 + ln] = c[v] + bias;
  }
}

// ---------------------------------------------------------------------------
// Launch
// ---------------------------------------------------------------------------
extern "C" void kernel_launch(void* const* d_in, const int* in_sizes, int n_in,
                              void* d_out, int out_size, void* d_ws, size_t ws_size,
                              hipStream_t stream) {
  const float* x  = (const float*)d_in[0];
  const float* W1 = (const float*)d_in[1];
  const float* b1 = (const float*)d_in[2];
  const float* W2 = (const float*)d_in[3];
  const float* b2 = (const float*)d_in[4];
  const float* W3 = (const float*)d_in[5];
  const float* b3 = (const float*)d_in[6];
  float* out = (float*)d_out;

  // Workspace layout (bytes)
  char* ws = (char*)d_ws;
  size_t offA   = 0;                                   // A packed bf16: 24 MiB
  size_t offB2  = offA  + (size_t)BATCH * DIN * 2;     // W2 packed bf16: 12 MiB
  size_t offB3  = offB2 + (size_t)DIN * HIDDEN * 2;    // W3 packed bf16: 0.5 MiB
  size_t offH   = offB3 + (size_t)HIDDEN * OUTPUT * 2; // h bf16: 4 MiB
  unsigned short* Apk  = (unsigned short*)(ws + offA);
  unsigned short* B2pk = (unsigned short*)(ws + offB2);
  unsigned short* B3pk = (unsigned short*)(ws + offB3);
  unsigned short* hbuf = (unsigned short*)(ws + offH);

  // 1) Pack weights into B-fragment layout
  {
    int total = (DIN >> 1) * HIDDEN;                   // 3,145,728
    pack_B_kernel<<<(total + 255) / 256, 256, 0, stream>>>(W2, (unsigned int*)B2pk, DIN, HIDDEN);
  }
  {
    int total = (HIDDEN >> 1) * OUTPUT;                // 131,072
    pack_B_kernel<<<(total + 255) / 256, 256, 0, stream>>>(W3, (unsigned int*)B3pk, HIDDEN, OUTPUT);
  }

  // 2) Fused conv + bf16 + A-fragment pack (streams the 192 MiB input once)
  {
    int total = BATCH * (DIN >> 1);                    // 6,291,456
    conv_pack_kernel<<<(total + 255) / 256, 256, 0, stream>>>(x, W1, b1, (unsigned int*)Apk);
  }

  // 3) GEMM1 + bias + ReLU  (grid: 32 N-tile-pairs x 32 M-tile-quads)
  gemm1_kernel<<<dim3(HIDDEN / 32, BATCH / 64), 32, 0, stream>>>(Apk, B2pk, b2, hbuf);

  // 4) GEMM2 + bias -> f32 output
  gemm2_kernel<<<dim3(OUTPUT / 16, BATCH / 16), 32, 0, stream>>>(hbuf, B3pk, b3, out);
}
